// biRNN_8031588843647
// MI455X (gfx1250) — compile-verified, matrix operand order checked by reading, not run on hardware
//
#include <hip/hip_runtime.h>
#include <hip/hip_bf16.h>
#include <hip/hip_fp16.h>

typedef __attribute__((ext_vector_type(16))) _Float16 v16h;
typedef __attribute__((ext_vector_type(8)))  float    v8f;

// Matches the builtin's parameter type exactly: 'int __attribute__((vector_size(16)))'
typedef int v4i_b128 __attribute__((vector_size(4 * sizeof(int))));
typedef __attribute__((address_space(1))) v4i_b128 g_b128;   // __device__ (global)
typedef __attribute__((address_space(3))) v4i_b128 l_b128;   // __shared__ (LDS)

constexpr int kB = 64;    // batch
constexpr int kS = 2048;  // seq len
constexpr int kI = 256;   // input dim
constexpr int kH = 256;   // hidden dim

// ---------------------------------------------------------------------------
// Phase A: parallel input projection for batch row 63 only (the only batch row
// the reference output reads).
//   pre[dir][t][h] = sum_i x[63][t][i] * Wih_dir[h][i] + b_ih_dir[h] + b_hh_dir[h]
// One wave computes a 16x16 (t x h) tile, chaining 8 v_wmma_f32_16x16x32_f16
// over K=256 with f32 accumulation.
// ---------------------------------------------------------------------------
__global__ void rnn_pregemm(const float* __restrict__ x,      // (B,S,I)
                            const float* __restrict__ w_ih,
                            const float* __restrict__ b_ih,
                            const float* __restrict__ b_hh,
                            const float* __restrict__ w_ih_r,
                            const float* __restrict__ b_ih_r,
                            const float* __restrict__ b_hh_r,
                            float* __restrict__ pre)          // [2][S][H]
{
    const int t0  = blockIdx.x * 16;
    const int h0  = blockIdx.y * 16;
    const int dir = blockIdx.z;

    const float* W  = dir ? w_ih_r : w_ih;
    const float* bi = dir ? b_ih_r : b_ih;
    const float* bh = dir ? b_hh_r : b_hh;
    float* out = pre + (size_t)dir * kS * kH;

    const int lane = threadIdx.x & 31;
    const int r    = lane & 15;
    const int hi   = lane >> 4;

    const float* xrow = x + (size_t)63 * kS * kI + (size_t)(t0 + r) * kI; // A row m=r
    const float* wrow = W + (size_t)(h0 + r) * kI;                        // B col n=r

    v8f c = {};
    #pragma unroll
    for (int k0 = 0; k0 < kI; k0 += 32) {
        const int oa = k0 + hi * 8;
        v16h a, b;
        #pragma unroll
        for (int e = 0; e < 8; ++e) {
            a[e]     = (_Float16)xrow[oa + e];
            a[e + 8] = (_Float16)xrow[oa + 16 + e];
            b[e]     = (_Float16)wrow[oa + e];
            b[e + 8] = (_Float16)wrow[oa + 16 + e];
        }
        // (neg_a, A, neg_b, B, c_mod, C, reuse_a, reuse_b)
        c = __builtin_amdgcn_wmma_f32_16x16x32_f16(false, a, false, b,
                                                   (short)0, c, false, false);
    }

    // C/D layout: lane r = column n (=h), VGPR v = row m (= t), +8 for hi lanes.
    const float bias = bi[h0 + r] + bh[h0 + r];
    #pragma unroll
    for (int v = 0; v < 8; ++v) {
        const int m = v + hi * 8;
        out[(size_t)(t0 + m) * kH + (h0 + r)] = c[v] + bias;
    }
}

// ---------------------------------------------------------------------------
// Phase B: sequential scan, one workgroup per direction (2 WGPs in parallel).
// W_hh (256 KB) lives entirely in gfx1250's 320 KB LDS, packed as
//   w4[k/4][j][0..3] = W[j][k .. k+3]
// Each 16B global chunk maps to a contiguous 16B LDS chunk, so the one-time
// fill is issued as global_load_async_to_lds_b128 (ASYNCcnt path). Per step
// each thread does coalesced ds_load_b128 of its weights and broadcast float4
// reads of h; tanh uses the native v_tanh_f32 TRANS op when available
// (shortens the 2048-step serial critical path).
// ---------------------------------------------------------------------------
__global__ void rnn_recur(const float* __restrict__ w_hh,
                          const float* __restrict__ w_hh_r,
                          const float* __restrict__ h0in,   // (2,B,H)
                          const float* __restrict__ pre,    // [2][S][H]
                          float* __restrict__ out)          // (2S, H)
{
    extern __shared__ float lds[];
    float* w4  = lds;                       // 64*256*4 = 65536 floats (256 KB)
    float* hb0 = lds + 64 * kH * 4;         // 16B aligned
    float* hb1 = hb0 + kH + 8;              // keep 16B alignment

    const int dir = blockIdx.x;
    const int j   = threadIdx.x;            // 0..255 -> output neuron
    const float* W     = dir ? w_hh_r : w_hh;
    const float* myPre = pre + (size_t)dir * kS * kH;

    // ---- one-time LDS fill of packed W_hh ----
#if __has_builtin(__builtin_amdgcn_global_load_async_to_lds_b128)
    // 16384 chunks of 16B: chunk c = (jj, g) copies W[jj][4g..4g+3]
    //   -> w4 + g*1024 + jj*4  (contiguous 16B)
    for (int c = j; c < kH * (kH / 4); c += blockDim.x) {
        const int jj = c >> 6;              // W row (output neuron)
        const int g  = c & 63;              // k-group
        __builtin_amdgcn_global_load_async_to_lds_b128(
            (g_b128*)(W + (size_t)jj * kI + g * 4),
            (l_b128*)(w4 + g * (kH * 4) + jj * 4),
            0, 0);
    }
#if __has_builtin(__builtin_amdgcn_s_wait_asynccnt)
    __builtin_amdgcn_s_wait_asynccnt(0);
#else
    asm volatile("s_wait_asynccnt 0" ::: "memory");
#endif
#else
    for (int idx = j; idx < kH * kH; idx += blockDim.x) {
        const int jj = idx >> 8;            // row (output neuron)
        const int k  = idx & 255;           // input index
        w4[(k >> 2) * (kH * 4) + jj * 4 + (k & 3)] = W[idx];
    }
#endif
    hb0[j] = h0in[(size_t)dir * kB * kH + (size_t)63 * kH + j];
    __syncthreads();

    float* cur = hb0;
    float* nxt = hb1;
    for (int s = 0; s < kS; ++s) {
        const int t = dir ? (kS - 1 - s) : s;   // bwd scan consumes reversed time
        // keep the next pre rows warm (global_prefetch_b8)
        if (s + 8 < kS) {
            const int tp = dir ? (kS - 1 - (s + 8)) : (s + 8);
            __builtin_prefetch(myPre + (size_t)tp * kH + j, 0, 1);
        }
        float acc = myPre[(size_t)t * kH + j];
        #pragma unroll 8
        for (int k4 = 0; k4 < kH / 4; ++k4) {
            const float4 w = ((const float4*)(w4 + k4 * (kH * 4)))[j]; // coalesced b128
            const float4 h = ((const float4*)cur)[k4];                 // broadcast b128
            acc += h.x * w.x + h.y * w.y + h.z * w.z + h.w * w.w;
        }
#if __has_builtin(__builtin_amdgcn_tanhf)
        const float nh = __builtin_amdgcn_tanhf(acc);   // native v_tanh_f32
#else
        const float nh = tanhf(acc);
#endif
        out[(size_t)(2 * s + dir) * kH + j] = nh;   // interleave [f_s, b_s]
        nxt[j] = nh;
        __syncthreads();
        float* tmp = cur; cur = nxt; nxt = tmp;
    }
}

// ---------------------------------------------------------------------------
// Phase C: second tuple output is h0 passed through unchanged.
// ---------------------------------------------------------------------------
__global__ void copy_h0(const float* __restrict__ src, float* __restrict__ dst, int n)
{
    const int i = blockIdx.x * blockDim.x + threadIdx.x;
    if (i < n) dst[i] = src[i];
}

extern "C" void kernel_launch(void* const* d_in, const int* in_sizes, int n_in,
                              void* d_out, int out_size, void* d_ws, size_t ws_size,
                              hipStream_t stream)
{
    const float* input  = (const float*)d_in[0];
    const float* h0     = (const float*)d_in[1];
    const float* w_ih   = (const float*)d_in[2];
    const float* w_hh   = (const float*)d_in[3];
    const float* b_ih   = (const float*)d_in[4];
    const float* b_hh   = (const float*)d_in[5];
    const float* w_ih_r = (const float*)d_in[6];
    const float* w_hh_r = (const float*)d_in[7];
    const float* b_ih_r = (const float*)d_in[8];
    const float* b_hh_r = (const float*)d_in[9];

    float* out = (float*)d_out;
    float* pre = (float*)d_ws;              // 2*S*H floats = 4 MB scratch

    dim3 gridA(kS / 16, kH / 16, 2);        // 4096 waves of WMMA tiles
    rnn_pregemm<<<gridA, 32, 0, stream>>>(input, w_ih, b_ih, b_hh,
                                          w_ih_r, b_ih_r, b_hh_r, pre);

    const size_t ldsBytes = (size_t)(64 * kH * 4 + 2 * (kH + 8)) * sizeof(float); // ~264 KB
    rnn_recur<<<2, kH, ldsBytes, stream>>>(w_hh, w_hh_r, h0, pre, out);

    copy_h0<<<(2 * kB * kH + 255) / 256, 256, 0, stream>>>(
        h0, out + (size_t)2 * kS * kH, 2 * kB * kH);
}
